// Emb_Layers_22110491640675
// MI455X (gfx1250) — compile-verified, matrix operand order checked by reading, not run on hardware
//
#include <hip/hip_runtime.h>
#include <cstdint>
#include <cstddef>

// ---------------------------------------------------------------------------
// Types for WMMA fragments (gfx1250 / CDNA5, wave32)
// ---------------------------------------------------------------------------
typedef __bf16       bf16x16 __attribute__((ext_vector_type(16)));
typedef float        v8f     __attribute__((ext_vector_type(8)));
typedef unsigned int u32x4   __attribute__((ext_vector_type(4)));

union Frag {
    bf16x16 v;
    u32x4   q[2];
};

__device__ __forceinline__ unsigned short f2bf(float f) {
    // round-to-nearest-even f32 -> bf16
    unsigned int u = __float_as_uint(f);
    u += 0x7FFFu + ((u >> 16) & 1u);
    return (unsigned short)(u >> 16);
}

// ---------------------------------------------------------------------------
// CSR build: histogram -> exclusive scan -> fill
// seg = dst * 8 + edge_type   (matches reference composite segment id)
// ---------------------------------------------------------------------------
__global__ void hist_kernel(const int* __restrict__ ei, const int* __restrict__ et,
                            int* __restrict__ cnt, int E) {
    int e = blockIdx.x * blockDim.x + threadIdx.x;
    if (e < E) {
        int dst = ei[E + e];
        atomicAdd(&cnt[dst * 8 + et[e]], 1);
    }
}

__global__ void scan1_kernel(const int* __restrict__ in, int* __restrict__ out,
                             int* __restrict__ bsums, int n) {
    __shared__ int sd[256];
    int tid  = threadIdx.x;
    int base = blockIdx.x * 1024 + tid * 4;
    int v0 = (base + 0 < n) ? in[base + 0] : 0;
    int v1 = (base + 1 < n) ? in[base + 1] : 0;
    int v2 = (base + 2 < n) ? in[base + 2] : 0;
    int v3 = (base + 3 < n) ? in[base + 3] : 0;
    int tot = v0 + v1 + v2 + v3;
    sd[tid] = tot;
    __syncthreads();
    for (int off = 1; off < 256; off <<= 1) {
        int tv = (tid >= off) ? sd[tid - off] : 0;
        __syncthreads();
        sd[tid] += tv;
        __syncthreads();
    }
    int excl = sd[tid] - tot;
    if (base + 0 < n) out[base + 0] = excl;
    if (base + 1 < n) out[base + 1] = excl + v0;
    if (base + 2 < n) out[base + 2] = excl + v0 + v1;
    if (base + 3 < n) out[base + 3] = excl + v0 + v1 + v2;
    if (tid == 255) bsums[blockIdx.x] = sd[255];
}

__global__ void scan2_kernel(int* __restrict__ bsums, int nb) {
    __shared__ int sd[1024];
    int tid = threadIdx.x;
    int v = (tid < nb) ? bsums[tid] : 0;
    sd[tid] = v;
    __syncthreads();
    for (int off = 1; off < 1024; off <<= 1) {
        int tv = (tid >= off) ? sd[tid - off] : 0;
        __syncthreads();
        sd[tid] += tv;
        __syncthreads();
    }
    if (tid < nb) bsums[tid] = sd[tid] - v;   // exclusive
}

__global__ void scan3_kernel(int* __restrict__ out, const int* __restrict__ bsums,
                             int n, int total) {
    int i = blockIdx.x * blockDim.x + threadIdx.x;
    if (i < n) out[i] += bsums[i >> 10];
    if (i == 0) out[n] = total;
}

__global__ void fill_kernel(const int* __restrict__ ei, const int* __restrict__ et,
                            const int* __restrict__ offs, int* __restrict__ cursor,
                            int* __restrict__ srcs, int E) {
    int e = blockIdx.x * blockDim.x + threadIdx.x;
    if (e < E) {
        int dst = ei[E + e];
        int seg = dst * 8 + et[e];
        int pos = offs[seg] + atomicAdd(&cursor[seg], 1);
        srcs[pos] = ei[e];
    }
}

// ---------------------------------------------------------------------------
// Weight prep: f32 [r][k][n] -> bf16 [r][n][k]  (B^T layout so each lane's
// 16 B-fragment elements are two contiguous 16B loads)
// ---------------------------------------------------------------------------
#define SZ_W1 (8 * 128 * 64)
#define SZ_R1 (128 * 64)
#define SZ_W2 (8 * 64 * 32)
#define SZ_R2 (64 * 32)

__global__ void prep_weights_kernel(const float* __restrict__ W1, const float* __restrict__ r1,
                                    const float* __restrict__ W2, const float* __restrict__ r2,
                                    unsigned short* __restrict__ W1t, unsigned short* __restrict__ r1t,
                                    unsigned short* __restrict__ W2t, unsigned short* __restrict__ r2t) {
    int i = blockIdx.x * blockDim.x + threadIdx.x;
    if (i < SZ_W1) {
        int r = i >> 13, rem = i & 8191;
        int n = rem >> 7, k = rem & 127;
        W1t[i] = f2bf(W1[(r << 13) + k * 64 + n]);
    } else if (i < SZ_W1 + SZ_R1) {
        int j = i - SZ_W1;
        int n = j >> 7, k = j & 127;
        r1t[j] = f2bf(r1[k * 64 + n]);
    } else if (i < SZ_W1 + SZ_R1 + SZ_W2) {
        int j = i - (SZ_W1 + SZ_R1);
        int r = j >> 11, n = (j >> 6) & 31, k = j & 63;
        W2t[j] = f2bf(W2[(r << 11) + k * 32 + n]);
    } else if (i < SZ_W1 + SZ_R1 + SZ_W2 + SZ_R2) {
        int j = i - (SZ_W1 + SZ_R1 + SZ_W2);
        int n = j >> 6, k = j & 63;
        r2t[j] = f2bf(r2[k * 32 + n]);
    }
}

// ---------------------------------------------------------------------------
// Fused RGCN layer: gather per-(node,rel) means into LDS (bf16), then
// 16x16x32 bf16 WMMA against pre-transposed weights; bias + activation.
//   Y[16 nodes, F_OUT] = act( bias + X@root + sum_r mean_r @ W_r )
// ---------------------------------------------------------------------------
template <int F_IN, int F_OUT, int ACT>
__global__ __launch_bounds__(128) void rgcn_layer_kernel(
    const float* __restrict__ X, const int* __restrict__ offs,
    const int* __restrict__ srcs, const unsigned short* __restrict__ Wt,
    const unsigned short* __restrict__ rootT, const float* __restrict__ bias,
    float* __restrict__ Y, int nNodes) {
    constexpr int FPT = F_IN / 8;   // features per thread in gather phase

    __shared__ unsigned short sX[16 * F_IN];
    __shared__ unsigned short sMean[8 * 16 * F_IN];

    const int t     = threadIdx.x;
    const int node0 = blockIdx.x * 16;
    const int m     = t >> 3;       // node within tile (0..15)
    const int g     = t & 7;        // feature group (8 threads per node)
    const int node  = node0 + m;
    const bool valid = node < nNodes;

    // ---- stage x tile (root-term A matrix) ----
    {
        const float* xp = X + (size_t)node * F_IN + g * FPT;
        int o = m * F_IN + g * FPT;
#pragma unroll
        for (int i = 0; i < FPT; i += 4) {
            float4 v = valid ? *(const float4*)(xp + i) : make_float4(0.f, 0.f, 0.f, 0.f);
            sX[o + i + 0] = f2bf(v.x);
            sX[o + i + 1] = f2bf(v.y);
            sX[o + i + 2] = f2bf(v.z);
            sX[o + i + 3] = f2bf(v.w);
        }
    }

    // ---- per-relation neighbor means (gather via CSR; L2-resident reads) ----
    for (int r = 0; r < 8; ++r) {
        float acc[FPT];
#pragma unroll
        for (int i = 0; i < FPT; ++i) acc[i] = 0.f;
        int cnt = 0;
        if (valid) {
            int s   = node * 8 + r;
            int beg = offs[s];
            int end = offs[s + 1];
            cnt = end - beg;
            for (int e = beg; e < end; ++e) {
                const float* xp = X + (size_t)srcs[e] * F_IN + g * FPT;
#pragma unroll
                for (int i = 0; i < FPT; i += 4) {
                    float4 v = *(const float4*)(xp + i);
                    acc[i + 0] += v.x;
                    acc[i + 1] += v.y;
                    acc[i + 2] += v.z;
                    acc[i + 3] += v.w;
                }
            }
        }
        float inv = (cnt > 0) ? 1.0f / (float)cnt : 0.0f;
        int o = r * 16 * F_IN + m * F_IN + g * FPT;
#pragma unroll
        for (int i = 0; i < FPT; ++i) sMean[o + i] = f2bf(acc[i] * inv);
    }
    __syncthreads();

    // ---- WMMA phase: each wave owns one 16-wide output column tile ----
    const int wave = t >> 5;
    const int lane = t & 31;
    constexpr int NT = F_OUT / 16;
    if (wave < NT) {
        // per-lane fragment addressing (16-bit A/B VGPR striping):
        // lanes 0-15: K = kb..kb+7 and kb+16..kb+23 ; lanes 16-31: +8
        const int arow = lane & 15;
        const int koff = (lane >> 4) << 3;
        const int nrow = wave * 16 + arow;

        v8f acc = {};
#pragma unroll 1
        for (int a = 0; a < 9; ++a) {
            const unsigned short* Asrc = (a == 0) ? &sX[0] : &sMean[(a - 1) * 16 * F_IN];
            const unsigned short* Bsrc = (a == 0) ? rootT : (Wt + (size_t)(a - 1) * F_OUT * F_IN);
#pragma unroll
            for (int kt = 0; kt < F_IN / 32; ++kt) {
                Frag af, bfr;
                const int kb = kt * 32 + koff;
                af.q[0]  = *(const u32x4*)(Asrc + arow * F_IN + kb);
                af.q[1]  = *(const u32x4*)(Asrc + arow * F_IN + kb + 16);
                bfr.q[0] = *(const u32x4*)(Bsrc + nrow * F_IN + kb);
                bfr.q[1] = *(const u32x4*)(Bsrc + nrow * F_IN + kb + 16);
                acc = __builtin_amdgcn_wmma_f32_16x16x32_bf16(
                    false, af.v, false, bfr.v, (short)0, acc, false, false);
            }
        }

        // ---- epilogue: C/D layout -> lane(0-15): M=j, lane(16-31): M=j+8 ----
        const int mrow0 = (lane >> 4) << 3;
        const int ncol  = wave * 16 + (lane & 15);
        const float bv  = bias[ncol];
        float* yp = Y + (size_t)(node0 + mrow0) * F_OUT + ncol;
        if (node0 + 16 <= nNodes) {
            // full tile (always the case when N % 16 == 0): unpredicated stores
#pragma unroll
            for (int j = 0; j < 8; ++j) {
                float v = acc[j] + bv;
                if (ACT == 0) v = fmaxf(v, 0.f);                    // relu
                else          v = 1.0f / (1.0f + __expf(-v));       // sigmoid
                yp[(size_t)j * F_OUT] = v;
            }
        } else {
#pragma unroll
            for (int j = 0; j < 8; ++j) {
                if (node0 + mrow0 + j < nNodes) {
                    float v = acc[j] + bv;
                    if (ACT == 0) v = fmaxf(v, 0.f);
                    else          v = 1.0f / (1.0f + __expf(-v));
                    yp[(size_t)j * F_OUT] = v;
                }
            }
        }
    }
}

// ---------------------------------------------------------------------------
// Host launcher
// ---------------------------------------------------------------------------
extern "C" void kernel_launch(void* const* d_in, const int* in_sizes, int n_in,
                              void* d_out, int out_size, void* d_ws, size_t ws_size,
                              hipStream_t stream) {
    (void)n_in; (void)out_size; (void)ws_size;
    const int*   ei = (const int*)d_in[0];     // edge_index (2,E): [0..E)=src, [E..2E)=dst
    const int*   et = (const int*)d_in[1];     // edge_type (E)
    const float* x0 = (const float*)d_in[2];   // embedding (N,128)
    const float* W1 = (const float*)d_in[3];
    const float* r1 = (const float*)d_in[4];
    const float* b1 = (const float*)d_in[5];
    const float* W2 = (const float*)d_in[6];
    const float* r2 = (const float*)d_in[7];
    const float* b2 = (const float*)d_in[8];
    float* out = (float*)d_out;

    const int E    = in_sizes[1];
    const int N    = in_sizes[2] / 128;
    const int NSEG = N * 8;
    const int nblk1 = (NSEG + 1023) / 1024;

    // workspace layout (256B aligned regions)
    char* base = (char*)d_ws;
    size_t off = 0;
    auto take = [&](size_t bytes) -> void* {
        void* r = base + off;
        off = (off + bytes + 255) & ~(size_t)255;
        return r;
    };
    int* cnt   = (int*)take((size_t)NSEG * 4);        // reused as cursor later
    int* offs  = (int*)take((size_t)(NSEG + 1) * 4);
    int* bsums = (int*)take(4096);
    int* srcs  = (int*)take((size_t)E * 4);
    unsigned short* w1t = (unsigned short*)take((size_t)SZ_W1 * 2);
    unsigned short* r1t = (unsigned short*)take((size_t)SZ_R1 * 2);
    unsigned short* w2t = (unsigned short*)take((size_t)SZ_W2 * 2);
    unsigned short* r2t = (unsigned short*)take((size_t)SZ_R2 * 2);
    float* h1 = (float*)take((size_t)N * 64 * 4);

    // ---- CSR build (shared by both layers) + weight prep ----
    hipMemsetAsync(cnt, 0, (size_t)NSEG * 4, stream);
    prep_weights_kernel<<<(SZ_W1 + SZ_R1 + SZ_W2 + SZ_R2 + 255) / 256, 256, 0, stream>>>(
        W1, r1, W2, r2, w1t, r1t, w2t, r2t);
    hist_kernel<<<(E + 255) / 256, 256, 0, stream>>>(ei, et, cnt, E);
    scan1_kernel<<<nblk1, 256, 0, stream>>>(cnt, offs, bsums, NSEG);
    scan2_kernel<<<1, 1024, 0, stream>>>(bsums, nblk1);
    scan3_kernel<<<(NSEG + 255) / 256, 256, 0, stream>>>(offs, bsums, NSEG, E);
    hipMemsetAsync(cnt, 0, (size_t)NSEG * 4, stream);  // cursor
    fill_kernel<<<(E + 255) / 256, 256, 0, stream>>>(ei, et, offs, cnt, srcs, E);

    // ---- layer 1: 128 -> 64, relu ----
    rgcn_layer_kernel<128, 64, 0><<<(N + 15) / 16, 128, 0, stream>>>(
        x0, offs, srcs, w1t, r1t, b1, h1, N);
    // ---- layer 2: 64 -> 32, sigmoid ----
    rgcn_layer_kernel<64, 32, 1><<<(N + 15) / 16, 128, 0, stream>>>(
        h1, offs, srcs, w2t, r2t, b2, out, N);
}
